// ModulatedResnetBlock_14912126451952
// MI455X (gfx1250) — matrix operand order repaired and checked by reading, not compiled, not run on hardware
//
#include <hip/hip_runtime.h>

typedef __attribute__((ext_vector_type(16))) _Float16     v16h;
typedef __attribute__((ext_vector_type(8)))  float        v8f;
typedef __attribute__((ext_vector_type(8)))  unsigned int v8u;
typedef __attribute__((ext_vector_type(4)))  unsigned int u32x4;

#define BB   4
#define CC   256
#define HH   128
#define WW   128
#define WDIM 512
#define LDST 40   // padded LDS row stride (f16 elems); 80B stride -> conflict-free, 16B aligned

union Frag { v16h h; v8u u; };

// ---------------------------------------------------------------------------
// Kernel 1: styles  s[layer][b][cin] = dot(w[b,:], style_w[cin,:]) + style_b[cin]
// ---------------------------------------------------------------------------
__global__ __launch_bounds__(256) void styles_kernel(
    const float* __restrict__ w,
    const float* __restrict__ sw1, const float* __restrict__ sb1,
    const float* __restrict__ sw2, const float* __restrict__ sb2,
    float* __restrict__ s)
{
    int idx   = blockIdx.x * 256 + threadIdx.x;   // 0..2047
    int cin   = idx & 255;
    int b     = (idx >> 8) & 3;
    int layer = idx >> 10;
    const float* sw = layer ? sw2 : sw1;
    const float* sb = layer ? sb2 : sb1;
    const float* wr  = w  + (size_t)b   * WDIM;
    const float* swr = sw + (size_t)cin * WDIM;
    float acc = sb[cin];
    for (int i = 0; i < WDIM; ++i) acc += wr[i] * swr[i];
    s[idx] = acc;
}

// ---------------------------------------------------------------------------
// Kernel 2: modulate + demodulate, write f16 weights as [layer][b][tap][cout][cin]
// ---------------------------------------------------------------------------
__global__ __launch_bounds__(256) void modw_kernel(
    const float* __restrict__ cw1, const float* __restrict__ cw2,
    const float* __restrict__ s,   // [2][4][256]
    _Float16* __restrict__ wmod)   // [2][4][9][256][256]
{
    int cout  = blockIdx.x;
    int b     = blockIdx.y;
    int layer = blockIdx.z;
    const float* cw = layer ? cw2 : cw1;
    float sv = s[((size_t)layer * BB + b) * CC + threadIdx.x];
    int cin = threadIdx.x;

    const float* row = cw + ((size_t)cout * CC + cin) * 9;
    float vals[9];
    float ss = 0.0f;
    #pragma unroll
    for (int t = 0; t < 9; ++t) { float v = row[t] * sv; vals[t] = v; ss += v * v; }

    __shared__ float red[256];
    red[threadIdx.x] = ss;
    __syncthreads();
    for (int off = 128; off > 0; off >>= 1) {
        if (threadIdx.x < off) red[threadIdx.x] += red[threadIdx.x + off];
        __syncthreads();
    }
    float sig = rsqrtf(red[0] + 1e-8f);

    _Float16* wout = wmod + (((size_t)layer * BB + b) * 9) * (CC * CC)
                          + (size_t)cout * CC + cin;
    #pragma unroll
    for (int t = 0; t < 9; ++t) wout[(size_t)t * (CC * CC)] = (_Float16)(vals[t] * sig);
}

// ---------------------------------------------------------------------------
// Kernel 3: implicit-GEMM 3x3 conv via 9 tap-GEMMs, WMMA f16 -> f32 accum.
//   SECOND=false: in = x (f32), out = relu(conv) as f16 (h buffer)
//   SECOND=true : in = h (f16), out = x + conv as f32 (final)
// Workgroup: 256 thr (8 waves) -> 128 Cout x 128 W (one image row).
// ---------------------------------------------------------------------------
template<bool SECOND>
__global__ __launch_bounds__(256) void conv_kernel(
    const float*    __restrict__ xin32,
    const _Float16* __restrict__ xin16,
    const _Float16* __restrict__ wmodL,   // [B][9][256][256] for this layer
    const float*    __restrict__ xres,
    _Float16*       __restrict__ out16,
    float*          __restrict__ out32)
{
    __shared__ __align__(16) _Float16 lA[128 * LDST];  // [cout][cin]
    __shared__ __align__(16) _Float16 lB[128 * LDST];  // [pixel][cin] (transposed)

    const int h     = blockIdx.x;        // row 0..127
    const int b     = blockIdx.y;        // batch
    const int mbase = blockIdx.z * 128;  // cout half

    const int tid  = threadIdx.x;
    const int lane = tid & 31;
    const int wave = tid >> 5;
    const int wm   = (wave & 1) * 64;    // wave M offset within 128
    const int wn   = (wave >> 1) * 32;   // wave N offset within 128
    const int l16  = lane & 15;
    const int hi   = lane >> 4;          // lane half (0/1)

    v8f acc[4][2];
    #pragma unroll
    for (int mt = 0; mt < 4; ++mt)
        #pragma unroll
        for (int nt = 0; nt < 2; ++nt)
            #pragma unroll
            for (int r = 0; r < 8; ++r) acc[mt][nt][r] = 0.0f;

    const _Float16* wb = wmodL + (size_t)b * 9 * CC * CC;

    for (int tap = 0; tap < 9; ++tap) {
        const int ky = tap / 3 - 1;
        const int kx = tap % 3 - 1;
        const int hy = h + ky;
        if (hy < 0 || hy >= HH) continue;   // uniform across block: whole tap is zero

        for (int cin0 = 0; cin0 < CC; cin0 += 32) {
            __syncthreads();   // previous iter's LDS reads complete

            // ---- stage A: weights [128 cout x 32 cin], 16B vector copies ----
            {
                const _Float16* wt = wb + ((size_t)tap * CC + mbase) * CC + cin0;
                #pragma unroll
                for (int c = 0; c < 2; ++c) {
                    int chunk = tid + c * 256;       // 0..511 (row x 4 chunks of 8 f16)
                    int row   = chunk >> 2;
                    int ch    = chunk & 3;
                    u32x4 d = *(const u32x4*)(wt + row * CC + ch * 8);
                    *(u32x4*)(&lA[row * LDST + ch * 8]) = d;
                }
            }
            // ---- stage B: input row, shifted by kx, zero-padded, transposed ----
            {
                const int ci = tid >> 3;            // 0..31 cin within slab
                const int p0 = (tid & 7) * 16;      // pixel chunk base
                if (!SECOND) {
                    const float* src = xin32 + ((size_t)(b * CC + cin0 + ci) * HH + hy) * WW;
                    #pragma unroll
                    for (int i = 0; i < 16; ++i) {
                        int wq = p0 + i + kx;
                        float v = (wq >= 0 && wq < WW) ? src[wq] : 0.0f;
                        lB[(p0 + i) * LDST + ci] = (_Float16)v;
                    }
                } else {
                    const _Float16* src = xin16 + ((size_t)(b * CC + cin0 + ci) * HH + hy) * WW;
                    #pragma unroll
                    for (int i = 0; i < 16; ++i) {
                        int wq = p0 + i + kx;
                        _Float16 v = (wq >= 0 && wq < WW) ? src[wq] : (_Float16)0.0f;
                        lB[(p0 + i) * LDST + ci] = v;
                    }
                }
            }
            __syncthreads();

            // ---- gather fragments in contiguous k order (k = 16*half + s), same for A and B ----
            Frag af[4], bf[2];
            #pragma unroll
            for (int mt = 0; mt < 4; ++mt) {
                const unsigned short* row =
                    (const unsigned short*)&lA[(wm + mt * 16 + l16) * LDST];
                #pragma unroll
                for (int v = 0; v < 8; ++v) {
                    // lanes 0-15: K = 2v,2v+1 ; lanes 16-31: K = 16+2v,17+2v
                    int koff = hi * 16 + v * 2;
                    af[mt].u[v] = *(const unsigned int*)(row + koff);
                }
            }
            #pragma unroll
            for (int nt = 0; nt < 2; ++nt) {
                const unsigned short* row =
                    (const unsigned short*)&lB[(wn + nt * 16 + l16) * LDST];
                #pragma unroll
                for (int v = 0; v < 8; ++v) {
                    // lanes 0-15: K = 2v,2v+1 ; lanes 16-31: K = 16+2v,17+2v
                    int koff = hi * 16 + v * 2;
                    bf[nt].u[v] = *(const unsigned int*)(row + koff);
                }
            }

            // ---- 8 WMMAs per wave per K-step ----
            #pragma unroll
            for (int mt = 0; mt < 4; ++mt)
                #pragma unroll
                for (int nt = 0; nt < 2; ++nt)
                    acc[mt][nt] = __builtin_amdgcn_wmma_f32_16x16x32_f16(
                        false, af[mt].h, false, bf[nt].h,
                        (short)0, acc[mt][nt], false, false);
        }
    }

    // ---- epilogue: C layout = lanes(0-15 N=l, VGPR r -> M=r ; lanes16-31 M=8+r) ----
    #pragma unroll
    for (int mt = 0; mt < 4; ++mt) {
        int coutb = mbase + wm + mt * 16 + hi * 8;
        #pragma unroll
        for (int nt = 0; nt < 2; ++nt) {
            int wpix = wn + nt * 16 + l16;
            #pragma unroll
            for (int r = 0; r < 8; ++r) {
                size_t o = (((size_t)b * CC + (coutb + r)) * HH + h) * WW + wpix;
                float v = acc[mt][nt][r];
                if (!SECOND) {
                    out16[o] = (_Float16)fmaxf(v, 0.0f);   // fused ReLU, f16 h
                } else {
                    out32[o] = xres[o] + v;                // fused residual add
                }
            }
        }
    }
}

// ---------------------------------------------------------------------------
extern "C" void kernel_launch(void* const* d_in, const int* in_sizes, int n_in,
                              void* d_out, int out_size, void* d_ws, size_t ws_size,
                              hipStream_t stream) {
    (void)in_sizes; (void)n_in; (void)out_size; (void)ws_size;
    const float* x   = (const float*)d_in[0];
    const float* w   = (const float*)d_in[1];
    const float* cw1 = (const float*)d_in[2];
    const float* cw2 = (const float*)d_in[3];
    const float* sw1 = (const float*)d_in[4];
    const float* sb1 = (const float*)d_in[5];
    const float* sw2 = (const float*)d_in[6];
    const float* sb2 = (const float*)d_in[7];
    float* out = (float*)d_out;

    char* ws = (char*)d_ws;
    float*    s     = (float*)ws;                               // 2*4*256 f32 = 8 KB
    _Float16* wmod  = (_Float16*)(ws + 8192);                   // 2*4*9*256*256 f16 = 9.4 MB
    _Float16* hbuf  = (_Float16*)(ws + 8192 + (size_t)2 * BB * 9 * CC * CC * sizeof(_Float16));

    styles_kernel<<<8, 256, 0, stream>>>(w, sw1, sb1, sw2, sb2, s);
    modw_kernel<<<dim3(CC, BB, 2), 256, 0, stream>>>(cw1, cw2, s, wmod);

    conv_kernel<false><<<dim3(HH, BB, 2), 256, 0, stream>>>(
        x, nullptr, wmod, nullptr, hbuf, nullptr);

    conv_kernel<true><<<dim3(HH, BB, 2), 256, 0, stream>>>(
        nullptr, hbuf, wmod + (size_t)BB * 9 * CC * CC, x, nullptr, out);
}
